// HPG_SSM_Core_76476187673191
// MI455X (gfx1250) — compile-verified, matrix operand order checked by reading, not run on hardware
//
#include <hip/hip_runtime.h>
#include <hip/hip_bf16.h>
#include <math.h>

// ---------- types for WMMA ----------
typedef __attribute__((ext_vector_type(16))) __bf16 v16bf;
typedef __attribute__((ext_vector_type(8)))  float v8f;
typedef __attribute__((ext_vector_type(8)))  unsigned short v8us;
typedef __attribute__((ext_vector_type(16))) unsigned short v16us;

// ---------- helpers ----------
__device__ __forceinline__ unsigned short f32_to_bf16_rne(float f) {
  union { float f; unsigned int u; } v; v.f = f;
  unsigned int u = v.u;
  if ((u & 0x7fffffffu) > 0x7f800000u)           // NaN -> quiet NaN
    return (unsigned short)((u >> 16) | 0x40u);
  return (unsigned short)((u + 0x7fffu + ((u >> 16) & 1u)) >> 16);
}

__device__ __forceinline__ float sigmoidf_fast(float x) {
  return 1.0f / (1.0f + __expf(-x));
}

// Load a 16-element bf16 fragment per lane: two contiguous 8-elem (16B) chunks.
// Matches CDNA5 16-bit A/B packing: elems[0..7] = K(half*8 .. +7),
// elems[8..15] = K(16+half*8 .. +7).
__device__ __forceinline__ v16bf load_frag16(const unsigned short* __restrict__ p) {
  v8us lo = *(const v8us*)(p);
  v8us hi = *(const v8us*)(p + 16);
  v16us c = __builtin_shufflevector(lo, hi, 0, 1, 2, 3, 4, 5, 6, 7,
                                            8, 9, 10, 11, 12, 13, 14, 15);
  return __builtin_bit_cast(v16bf, c);
}

// ---------- generic bf16 WMMA GEMM: C[M,N] = A[M,K] * Bw[N,K]^T ----------
// K must be a multiple of 64 (true for all call sites: 64, 1024, 2048).
// MODE 0: Cf = acc
// MODE 1: Cbf = bf16( aux[m,n] * sigmoid(acc + bias[n]) )     (gate-modulate)
// MODE 2: Cf  = softplus(acc + bias[n])                        (dt path)
template <int NSUB, int MODE>
__global__ __launch_bounds__(32)
void gemm_bf16_wmma(const unsigned short* __restrict__ A,
                    const unsigned short* __restrict__ Bw,
                    int N, int K, int n_base,
                    float* __restrict__ Cf,
                    unsigned short* __restrict__ Cbf,
                    const float* __restrict__ bias,
                    const float* __restrict__ aux)
{
  const int lane   = threadIdx.x & 31;
  const int row    = lane & 15;
  const int half   = lane >> 4;
  const int tileM  = blockIdx.y << 4;
  const int tileN0 = n_base + (blockIdx.x << 6);

  v8f acc[NSUB];
#pragma unroll
  for (int j = 0; j < NSUB; ++j)
#pragma unroll
    for (int r = 0; r < 8; ++r) acc[j][r] = 0.0f;

  const unsigned short* aRow = A + (size_t)(tileM + row) * K + half * 8;
  const unsigned short* bRow[NSUB];
#pragma unroll
  for (int j = 0; j < NSUB; ++j)
    bRow[j] = Bw + (size_t)(tileN0 + j * 16 + row) * K + half * 8;

  // Two-stage K loop: issue all loads for k and k+32 up front, then the two
  // WMMA groups, so the second stage's loads stay in flight behind the first
  // stage's matrix ops (partial s_wait_loadcnt instead of full drains).
  for (int k = 0; k < K; k += 64) {
    v16bf a0 = load_frag16(aRow + k);
    v16bf b0[NSUB];
#pragma unroll
    for (int j = 0; j < NSUB; ++j) b0[j] = load_frag16(bRow[j] + k);

    v16bf a1 = load_frag16(aRow + k + 32);
    v16bf b1[NSUB];
#pragma unroll
    for (int j = 0; j < NSUB; ++j) b1[j] = load_frag16(bRow[j] + k + 32);

#pragma unroll
    for (int j = 0; j < NSUB; ++j)
      acc[j] = __builtin_amdgcn_wmma_f32_16x16x32_bf16(
          false, a0, false, b0[j], (short)0, acc[j], false, false);
#pragma unroll
    for (int j = 0; j < NSUB; ++j)
      acc[j] = __builtin_amdgcn_wmma_f32_16x16x32_bf16(
          false, a1, false, b1[j], (short)0, acc[j], false, false);
  }

  // epilogue: C layout -> VGPR r: m = tileM + half*8 + r, n = tileN0 + j*16 + row
#pragma unroll
  for (int j = 0; j < NSUB; ++j) {
    const int n = tileN0 + j * 16 + row;
    const float bval = (MODE != 0) ? bias[n] : 0.0f;
#pragma unroll
    for (int r = 0; r < 8; ++r) {
      const int m = tileM + half * 8 + r;
      const size_t idx = (size_t)m * N + n;
      float v = acc[j][r];
      if (MODE == 0) {
        Cf[idx] = v;
      } else if (MODE == 1) {
        float g = sigmoidf_fast(v + bval);
        Cbf[idx] = f32_to_bf16_rne(aux[idx] * g);
      } else {
        float s = v + bval;
        s = (s > 20.0f) ? s : log1pf(__expf(s));
        Cf[idx] = s;
      }
    }
  }
}

// ---------- causal depthwise conv (taps=4) + bias + SiLU ----------
// x_in = cols [0,2048) of the (token,4096) xz buffer
__global__ __launch_bounds__(256)
void conv_silu_kernel(const float* __restrict__ xz,
                      const float* __restrict__ cw,   // (2048,1,4)
                      const float* __restrict__ cb,   // (2048)
                      float* __restrict__ xcf,
                      unsigned short* __restrict__ xcb)
{
  long long idx = (long long)blockIdx.x * blockDim.x + threadIdx.x;
  const long long total = 4096LL * 2048LL;
  if (idx >= total) return;
  const int c = (int)(idx & 2047);
  const long long t = idx >> 11;          // token = b*L + l
  const int l = (int)(t & 2047);

  const float w0 = cw[c * 4 + 0], w1 = cw[c * 4 + 1];
  const float w2 = cw[c * 4 + 2], w3 = cw[c * 4 + 3];
  const float* base = xz + t * 4096 + c;

  float acc = w3 * base[0];               // out[l] = sum_k w[k]*x[l-3+k]
  if (l >= 1) acc += w2 * base[-4096];
  if (l >= 2) acc += w1 * base[-2 * 4096];
  if (l >= 3) acc += w0 * base[-3 * 4096];

  float v = acc + cb[c];
  float s = v * sigmoidf_fast(v);
  xcf[idx] = s;
  xcb[idx] = f32_to_bf16_rne(s);
}

// ---------- extract dt_r (first 64 cols of x_dbl) as bf16 ----------
__global__ __launch_bounds__(256)
void extract_dtr_kernel(const float* __restrict__ xdbl, unsigned short* __restrict__ dtr)
{
  int i = blockIdx.x * blockDim.x + threadIdx.x;
  if (i >= 4096 * 64) return;
  int r = i >> 6, c = i & 63;
  dtr[i] = f32_to_bf16_rne(xdbl[r * 96 + c]);
}

// ---------- selective-scan recurrence ----------
// one thread per (b, channel); 16 states in registers; A = -exp(A_log)
__global__ __launch_bounds__(256)
void ssm_scan_kernel(const float* __restrict__ dt,    // (token,2048)
                     const float* __restrict__ xdbl,  // (token,96): [64:80)=B, [80:96)=C
                     const float* __restrict__ xc,    // (token,2048)
                     const float* __restrict__ A_log, // (2048,16)
                     float* __restrict__ ys)          // (token,2048)
{
  const int gid = blockIdx.x * blockDim.x + threadIdx.x;   // 0..4095
  const int b = gid >> 11;
  const int e = gid & 2047;

  float a[16], h[16];
#pragma unroll
  for (int s = 0; s < 16; ++s) { a[s] = -__expf(A_log[e * 16 + s]); h[s] = 0.0f; }

  for (int l = 0; l < 2048; ++l) {
    const long long tok = (long long)b * 2048 + l;
    const float dtv = dt[tok * 2048 + e];
    const float xv  = xc[tok * 2048 + e];
    const float* bc = xdbl + tok * 96 + 64;   // broadcast across block
    const float dtx = dtv * xv;
    float y = 0.0f;
#pragma unroll
    for (int s = 0; s < 16; ++s) {
      const float dA = __expf(dtv * a[s]);
      h[s] = dA * h[s] + dtx * bc[s];
      y += h[s] * bc[16 + s];
    }
    ys[tok * 2048 + e] = y;
  }
}

// ---------- y = (ys + D*x_c) * silu(z), store bf16 ----------
__global__ __launch_bounds__(256)
void ymix_kernel(const float* __restrict__ ys,
                 const float* __restrict__ xcf,
                 const float* __restrict__ xz,
                 const float* __restrict__ Dv,
                 unsigned short* __restrict__ ybf)
{
  long long idx = (long long)blockIdx.x * blockDim.x + threadIdx.x;
  if (idx >= 4096LL * 2048LL) return;
  const int e = (int)(idx & 2047);
  const long long t = idx >> 11;
  const float z = xz[t * 4096 + 2048 + e];
  const float v = (ys[idx] + Dv[e] * xcf[idx]) * (z * sigmoidf_fast(z));
  ybf[idx] = f32_to_bf16_rne(v);
}

// ---------- f32 -> bf16 bulk convert ----------
__global__ __launch_bounds__(256)
void cvt_f32_bf16_kernel(const float* __restrict__ src, unsigned short* __restrict__ dst,
                         long long n)
{
  long long i = (long long)blockIdx.x * blockDim.x + threadIdx.x;
  if (i < n) dst[i] = f32_to_bf16_rne(src[i]);
}

// ---------- host launcher ----------
extern "C" void kernel_launch(void* const* d_in, const int* in_sizes, int n_in,
                              void* d_out, int out_size, void* d_ws, size_t ws_size,
                              hipStream_t stream)
{
  (void)in_sizes; (void)n_in; (void)out_size; (void)ws_size;

  const float* x_f    = (const float*)d_in[0];
  const float* x_h    = (const float*)d_in[1];
  const float* W_hf   = (const float*)d_in[2];
  const float* b_hf   = (const float*)d_in[3];
  const float* W_in   = (const float*)d_in[4];
  const float* conv_w = (const float*)d_in[5];
  const float* conv_b = (const float*)d_in[6];
  const float* W_x    = (const float*)d_in[7];
  const float* W_dt   = (const float*)d_in[8];
  const float* b_dt   = (const float*)d_in[9];
  const float* A_log  = (const float*)d_in[10];
  const float* Dvec   = (const float*)d_in[11];
  const float* W_out  = (const float*)d_in[12];
  float* out = (float*)d_out;

  const int M = 4096;                    // B*L tokens
  const int DM = 1024, DI = 2048;
  const int MT = M / 16;                 // 256 M-tiles

  char* ws = (char*)d_ws;
  size_t off = 0;
  auto alloc = [&](size_t bytes) -> void* {
    void* p = ws + off;
    off = (off + bytes + 255) & ~(size_t)255;
    return p;
  };

  unsigned short* xh_bf   = (unsigned short*)alloc((size_t)M * DM * 2);
  unsigned short* Whf_bf  = (unsigned short*)alloc((size_t)DM * DM * 2);
  unsigned short* Win_bf  = (unsigned short*)alloc((size_t)2 * DI * DM * 2);
  unsigned short* Wx_bf   = (unsigned short*)alloc((size_t)96 * DI * 2);
  unsigned short* Wdt_bf  = (unsigned short*)alloc((size_t)DI * 64 * 2);
  unsigned short* Wout_bf = (unsigned short*)alloc((size_t)DM * DI * 2);
  unsigned short* u_bf    = (unsigned short*)alloc((size_t)M * DM * 2);
  float*          xz      = (float*)alloc((size_t)M * 2 * DI * 4);
  float*          xcf     = (float*)alloc((size_t)M * DI * 4);
  unsigned short* xcb     = (unsigned short*)alloc((size_t)M * DI * 2);
  float*          xdbl    = (float*)alloc((size_t)M * 96 * 4);
  unsigned short* dtr     = (unsigned short*)alloc((size_t)M * 64 * 2);
  float*          dtb     = (float*)alloc((size_t)M * DI * 4);
  float*          ysb     = (float*)alloc((size_t)M * DI * 4);
  unsigned short* ybf     = (unsigned short*)alloc((size_t)M * DI * 2);

  auto cvt = [&](const float* s, unsigned short* d, long long n) {
    int blocks = (int)((n + 255) / 256);
    cvt_f32_bf16_kernel<<<blocks, 256, 0, stream>>>(s, d, n);
  };

  // weight / activation conversions
  cvt(x_h,   xh_bf,   (long long)M * DM);
  cvt(W_hf,  Whf_bf,  (long long)DM * DM);
  cvt(W_in,  Win_bf,  (long long)2 * DI * DM);
  cvt(W_x,   Wx_bf,   (long long)96 * DI);
  cvt(W_dt,  Wdt_bf,  (long long)DI * 64);
  cvt(W_out, Wout_bf, (long long)DM * DI);

  // G1: u = x_f * sigmoid(x_h @ W_hf^T + b_hf)   -> bf16
  gemm_bf16_wmma<4, 1><<<dim3(DM / 64, MT), 32, 0, stream>>>(
      xh_bf, Whf_bf, DM, DM, 0, nullptr, u_bf, b_hf, x_f);

  // G2: xz = u @ W_in^T  (token, 4096) f32
  gemm_bf16_wmma<4, 0><<<dim3((2 * DI) / 64, MT), 32, 0, stream>>>(
      u_bf, Win_bf, 2 * DI, DM, 0, xz, nullptr, nullptr, nullptr);

  // conv + SiLU -> x_c (f32 + bf16)
  {
    long long total = (long long)M * DI;
    conv_silu_kernel<<<(int)((total + 255) / 256), 256, 0, stream>>>(
        xz, conv_w, conv_b, xcf, xcb);
  }

  // G3: x_dbl = x_c @ W_x^T  (token, 96) f32: cols 0..63 (NSUB=4) + 64..95 (NSUB=2)
  gemm_bf16_wmma<4, 0><<<dim3(1, MT), 32, 0, stream>>>(
      xcb, Wx_bf, 96, DI, 0, xdbl, nullptr, nullptr, nullptr);
  gemm_bf16_wmma<2, 0><<<dim3(1, MT), 32, 0, stream>>>(
      xcb, Wx_bf, 96, DI, 64, xdbl, nullptr, nullptr, nullptr);

  // dt_r -> bf16
  extract_dtr_kernel<<<(M * 64 + 255) / 256, 256, 0, stream>>>(xdbl, dtr);

  // G4: dt = softplus(dt_r @ W_dt^T + b_dt)  (token, 2048) f32
  gemm_bf16_wmma<4, 2><<<dim3(DI / 64, MT), 32, 0, stream>>>(
      dtr, Wdt_bf, DI, 64, 0, dtb, nullptr, b_dt, nullptr);

  // recurrent scan
  ssm_scan_kernel<<<16, 256, 0, stream>>>(dtb, xdbl, xcf, A_log, ysb);

  // y = (ys + D*x_c) * silu(z) -> bf16
  {
    long long total = (long long)M * DI;
    ymix_kernel<<<(int)((total + 255) / 256), 256, 0, stream>>>(ysb, xcf, xz, Dvec, ybf);
  }

  // G5: out = y @ W_out^T  (token, 1024) f32
  gemm_bf16_wmma<4, 0><<<dim3(DM / 64, MT), 32, 0, stream>>>(
      ybf, Wout_bf, DM, DI, 0, out, nullptr, nullptr, nullptr);
}
